// KimiDeltaAttention_66056597012601
// MI455X (gfx1250) — matrix operand order repaired and checked by reading, compiled, not run on hardware
//
#include <hip/hip_runtime.h>
#include <math.h>

// ---------------------------------------------------------------------------
// Kimi Delta Attention on gfx1250 (MI455X).
// Big projections: bf16-native WMMA GEMM (v_wmma_f32_16x16x32_bf16):
//   operands pre-converted to bf16 once (halves L2/HBM traffic), 128x64 block
//   tile, BK=64, double-buffered LDS, reg-prefetch + global_prefetch_b8.
// Chunked delta-rule scan: fp32 WMMA (v_wmma_f32_16x16x4_f32), state in LDS.
// ---------------------------------------------------------------------------

#define D_MODEL 2048
#define NHEADS  16
#define DKH     128
#define DVH     128
#define CHUNKT  64

typedef __attribute__((ext_vector_type(16))) __bf16        v16bf;
typedef __attribute__((ext_vector_type(8)))  __bf16        bf16x8;
typedef __attribute__((ext_vector_type(4)))  __bf16        bf16x4;
typedef __attribute__((ext_vector_type(8)))  float         v8f;
typedef __attribute__((ext_vector_type(2)))  float         v2f;
typedef __attribute__((ext_vector_type(4)))  unsigned int  u32x4;

__device__ __forceinline__ float sigmoid_(float x) { return 1.f / (1.f + __expf(-x)); }

// ===========================================================================
// fp32 -> bf16 conversion (b128 in, b64 out). Requires n % 4 == 0.
// ===========================================================================
__global__ void to_bf16(const float* __restrict__ in, __bf16* __restrict__ out, int n)
{
    int i = (blockIdx.x * blockDim.x + threadIdx.x) * 4;
    if (i < n) {
        float4 f = *(const float4*)&in[i];
        bf16x4 h = { (__bf16)f.x, (__bf16)f.y, (__bf16)f.z, (__bf16)f.w };
        *(bf16x4*)&out[i] = h;
    }
}

// ===========================================================================
// GEMM: C[M,N] = A[M,K] * B[K,N]  (A,B bf16 in HBM; fp32 accum; fp32 C)
// Block tile 128x64, BK=64, 256 threads = 8 waves; wave w owns rows
// [16w,16w+16) x all 64 cols -> 4 accumulators, 8 WMMAs per K-stage.
// Double-buffered LDS + register prefetch: 1 barrier/stage; prefetch hints
// two stages ahead. NG=true adds N-guards (beta GEMM, N=16).
// Requires: M%128==0, K%64==0, and (!NG -> N%64==0).
// ===========================================================================
#define BM 128
#define BN 64
#define BKs 64

template <bool NG>
__global__ __launch_bounds__(256)
void gemm_bf16_wmma(const __bf16* __restrict__ A, const __bf16* __restrict__ B,
                    float* __restrict__ C, int M, int N, int K)
{
    __shared__ __attribute__((aligned(64))) __bf16 sA[2][BM][BKs];  // [m][k]
    __shared__ __attribute__((aligned(64))) __bf16 sB[2][BN][BKs];  // [n][k] (B^T)

    const int tid  = threadIdx.x;
    const int lane = tid & 31;
    const int wave = tid >> 5;                 // 0..7 -> 16-row slab
    const int blockM = blockIdx.y * BM;
    const int blockN = blockIdx.x * BN;

    v8f acc[4] = {{}, {}, {}, {}};
    bf16x8 pa[4];                              // A prefetch: 128x64 bf16 / 256thr
    bf16x8 pb[2];                              // B prefetch: 64x64 bf16 / 256thr

    auto fetch = [&](int k0) {
        #pragma unroll
        for (int i = 0; i < 4; ++i) {
            int v   = tid + i * 256;           // b128 chunk index, 0..1023
            int row = v >> 3;                  // 8 chunks per 64-wide row
            int col = (v & 7) * 8;
            pa[i] = *(const bf16x8*)&A[(size_t)(blockM + row) * K + k0 + col];
        }
        #pragma unroll
        for (int i = 0; i < 2; ++i) {
            int v  = tid + i * 256;            // 0..511
            int kr = v >> 3;
            int cn = (v & 7) * 8;
            if (NG) {
                bf16x8 t;
                #pragma unroll
                for (int e = 0; e < 8; ++e)
                    t[e] = (blockN + cn + e < N)
                         ? B[(size_t)(k0 + kr) * N + blockN + cn + e] : (__bf16)0.f;
                pb[i] = t;
            } else {
                pb[i] = *(const bf16x8*)&B[(size_t)(k0 + kr) * N + blockN + cn];
            }
        }
    };
    auto stash = [&](int buf) {
        #pragma unroll
        for (int i = 0; i < 4; ++i) {
            int v   = tid + i * 256;
            int row = v >> 3;
            int col = (v & 7) * 8;
            *(bf16x8*)&sA[buf][row][col] = pa[i];
        }
        #pragma unroll
        for (int i = 0; i < 2; ++i) {
            int v  = tid + i * 256;
            int kr = v >> 3;
            int cn = (v & 7) * 8;
            #pragma unroll
            for (int e = 0; e < 8; ++e)
                sB[buf][cn + e][kr] = pb[i][e];
        }
    };

    const int m_lo = wave * 16 + (lane & 15);  // A-frag row
    const int kb8  = (lane >> 4) * 8;          // unused: A frag also uses kb16
    const int kb16 = (lane >> 4) * 16;         // 16-bit B-frag K base
    const int nn   = lane & 15;

    fetch(0);
    stash(0);
    __syncthreads();

    const int nk = K / BKs;
    for (int kt = 0; kt < nk; ++kt) {
        const int cur = kt & 1;
        if (kt + 1 < nk) fetch((kt + 1) * BKs);    // overlap with WMMA below
        if (kt + 2 < nk) {                          // L2 prefetch 2 stages ahead
            const int k2 = (kt + 2) * BKs;
            __builtin_prefetch(&A[(size_t)(blockM + (tid >> 3)) * K + k2 + (tid & 7) * 8], 0, 3);
            __builtin_prefetch(&B[(size_t)(k2 + (tid >> 3)) * N + blockN + (tid & 7) * 8], 0, 3);
        }

        #pragma unroll
        for (int ks = 0; ks < BKs; ks += 32) {
            union { v16bf v; u32x4 q[2]; } af;
            af.q[0] = *(const u32x4*)&sA[cur][m_lo][ks + kb16];
            af.q[1] = *(const u32x4*)&sA[cur][m_lo][ks + kb16 + 8];
            #pragma unroll
            for (int j = 0; j < 4; ++j) {
                v16bf bfv = *(const v16bf*)&sB[cur][j * 16 + nn][ks + kb16];
                acc[j] = __builtin_amdgcn_wmma_f32_16x16x32_bf16(
                    false, af.v, false, bfv, (short)0, acc[j], false, false);
            }
        }

        if (kt + 1 < nk) stash(cur ^ 1);           // other buffer: no hazard
        __syncthreads();
    }

    // C/D layout: lane n = l&15; VGPR r -> m = r + 8*(l>>4)
    const int n0    = blockN + nn;
    const int mrow0 = blockM + wave * 16 + (lane >> 4) * 8;
    #pragma unroll
    for (int j = 0; j < 4; ++j) {
        const int gn = n0 + j * 16;
        if (NG && gn >= N) continue;
        #pragma unroll
        for (int r = 0; r < 8; ++r)
            C[(size_t)(mrow0 + r) * N + gn] = acc[j][r];
    }
}

// ===========================================================================
// Causal depthwise conv (K=4) + SiLU (+ optional per-head L2 norm).
// One wave per (b,t,h); lane owns 4 contiguous channels -> b128 loads/stores.
// ===========================================================================
__global__ __launch_bounds__(256)
void conv_silu_norm(const float* __restrict__ pre, const float* __restrict__ w,
                    float* __restrict__ out, int Bsz, int T, int do_l2)
{
    const int gw   = (blockIdx.x * blockDim.x + threadIdx.x) >> 5;
    const int lane = threadIdx.x & 31;
    if (gw >= Bsz * T * NHEADS) return;
    const int h  = gw % NHEADS;
    const int bt = gw / NHEADS;
    const int t  = bt % T;
    const int b  = bt / T;
    const int cch0 = h * DKH + lane * 4;

    float4 wv[4];
    #pragma unroll
    for (int e = 0; e < 4; ++e)
        wv[e] = *(const float4*)&w[(cch0 + e) * 4];   // 4 taps of channel e

    float acc[4] = {0.f, 0.f, 0.f, 0.f};
    #pragma unroll
    for (int j = 0; j < 4; ++j) {
        const int ts = t + j - 3;
        if (ts >= 0) {                                 // uniform across the wave
            float4 xv = *(const float4*)&pre[((size_t)b * T + ts) * D_MODEL + cch0];
            const float* xf = (const float*)&xv;
            #pragma unroll
            for (int e = 0; e < 4; ++e)
                acc[e] = fmaf(((const float*)&wv[e])[j], xf[e], acc[e]);
        }
    }

    float vals[4];
    float ss = 0.f;
    #pragma unroll
    for (int e = 0; e < 4; ++e) {
        vals[e] = acc[e] * sigmoid_(acc[e]);           // SiLU
        ss += vals[e] * vals[e];
    }
    if (do_l2) {
        #pragma unroll
        for (int off = 16; off > 0; off >>= 1) ss += __shfl_xor(ss, off, 32);
        const float scale = 1.f / fmaxf(sqrtf(ss), 1e-6f);
        #pragma unroll
        for (int e = 0; e < 4; ++e) vals[e] *= scale;
    }
    float4 ov = {vals[0], vals[1], vals[2], vals[3]};
    *(float4*)&out[((size_t)b * T + t) * D_MODEL + cch0] = ov;
}

// ===========================================================================
// Elementwise: g = -softplus(clip(x,-10,10));  beta = sigmoid(clip)
// ===========================================================================
__global__ void ew_neg_softplus(float* __restrict__ p, int n)
{
    int i = blockIdx.x * blockDim.x + threadIdx.x;
    if (i < n) {
        float x = fminf(fmaxf(p[i], -10.f), 10.f);
        p[i] = -log1pf(__expf(x));
    }
}
__global__ void ew_sigmoid_clip(float* __restrict__ p, int n)
{
    int i = blockIdx.x * blockDim.x + threadIdx.x;
    if (i < n) {
        float x = fminf(fmaxf(p[i], -10.f), 10.f);
        p[i] = sigmoid_(x);
    }
}

// ===========================================================================
// Chunked KDA scan. One workgroup (256 thr = 8 waves) per (b,h).
// State S[128][128] fp32 lives in LDS; all matmuls use v_wmma_f32_16x16x4_f32.
// Fragment layouts (ISA 7.12.2, 32-bit): A: m=l&15, k0=(l>>4)*2
//                                        B: n=l&15, b={B[k0][n],B[k0+1][n]}
//                                        C: n=l&15, vgpr r -> m=r+8*(l>>4)
// ===========================================================================
__device__ __forceinline__ void kda_qk_mm(const float* __restrict__ L,   // 64 x DKH
                                          const float* __restrict__ Kb,  // 64 x DKH
                                          float* __restrict__ Out,       // 64 x 64
                                          const float* __restrict__ betaArr,
                                          int lane, int wave)
{
    const int i  = wave >> 1;
    const int jb = (wave & 1) * 2;
    const int mrow = i * 16 + (lane & 15);
    #pragma unroll
    for (int jj = 0; jj < 2; ++jj) {
        const int j = jb + jj;
        const int ncol = j * 16 + (lane & 15);
        v8f acc = {};
        for (int kk = 0; kk < DKH; kk += 4) {
            const int k2 = kk + (lane >> 4) * 2;
            v2f a, bq;
            a.x  = L[mrow * DKH + k2];   a.y  = L[mrow * DKH + k2 + 1];
            bq.x = Kb[ncol * DKH + k2];  bq.y = Kb[ncol * DKH + k2 + 1];
            acc = __builtin_amdgcn_wmma_f32_16x16x4_f32(false, a, false, bq,
                                                        (short)0, acc, false, false);
        }
        #pragma unroll
        for (int r = 0; r < 8; ++r) {
            const int tt = i * 16 + (lane >> 4) * 8 + r;
            const int sscol = j * 16 + (lane & 15);
            float val;
            if (betaArr) val = (sscol < tt)  ? acc[r] * betaArr[tt] : 0.f;
            else         val = (sscol <= tt) ? acc[r]               : 0.f;
            Out[tt * CHUNKT + sscol] = val;
        }
    }
}

__global__ __launch_bounds__(256)
void kda_chunk_scan(const float* __restrict__ q, const float* __restrict__ k,
                    const float* __restrict__ v, const float* __restrict__ g,
                    const float* __restrict__ beta, float* __restrict__ o,
                    int Bsz, int T)
{
    extern __shared__ float smem[];
    float* sS    = smem;                    // 128*128
    float* sQt   = sS   + DKH * DVH;        // 64*128
    float* sKt   = sQt  + CHUNKT * DKH;     // 64*128
    float* sKb   = sKt  + CHUNKT * DKH;     // 64*128
    float* sU    = sKb  + CHUNKT * DKH;     // 64*128 (rhs -> u in place)
    float* sA    = sU   + CHUNKT * DVH;     // 64*64  (M, then attn)
    float* sEb   = sA   + CHUNKT * CHUNKT;  // 128    exp(b_last)
    float* sBeta = sEb  + DKH;              // 64

    const int tid  = threadIdx.x;
    const int lane = tid & 31;
    const int wave = tid >> 5;
    const int b = blockIdx.x / NHEADS;
    const int h = blockIdx.x % NHEADS;

    for (int i = tid; i < DKH * DVH; i += 256) sS[i] = 0.f;
    __syncthreads();

    const int nchunks = T / CHUNKT;
    for (int c = 0; c < nchunks; ++c) {
        const int t0 = c * CHUNKT;

        // -- stage 1: cumsum of g along t; build q_t, k_t, k_b; eb_last --
        if (tid < DKH) {
            const int d = tid;
            float cum = 0.f;
            for (int t = 0; t < CHUNKT; ++t) {
                const size_t idx = ((size_t)(b * T + t0 + t) * NHEADS + h) * DKH + d;
                cum += g[idx];
                const float e  = __expf(cum);
                const float ei = __expf(-cum);
                sQt[t * DKH + d] = q[idx] * e;
                sKt[t * DKH + d] = k[idx] * e;
                sKb[t * DKH + d] = k[idx] * ei;
            }
            sEb[d] = __expf(cum);
        }
        if (tid < CHUNKT)
            sBeta[tid] = beta[(size_t)(b * T + t0 + tid) * NHEADS + h];
        __syncthreads();

        // -- stage 2: M = beta .* strict_tril(Kt @ Kb^T)  -> sA --
        kda_qk_mm(sKt, sKb, sA, sBeta, lane, wave);

        // -- stage 3: rhs = beta .* (V - Kt @ S) -> sU --
        #pragma unroll
        for (int it = 0; it < 4; ++it) {
            const int tix = wave * 4 + it;
            const int ti = tix >> 3, tj = tix & 7;
            const int mrow = ti * 16 + (lane & 15);
            const int col  = tj * 16 + (lane & 15);
            v8f acc = {};
            for (int kk = 0; kk < DKH; kk += 4) {
                const int k2 = kk + (lane >> 4) * 2;
                v2f a, bq;
                a.x  = sKt[mrow * DKH + k2];  a.y  = sKt[mrow * DKH + k2 + 1];
                bq.x = sS[k2 * DVH + col];    bq.y = sS[(k2 + 1) * DVH + col];
                acc = __builtin_amdgcn_wmma_f32_16x16x4_f32(false, a, false, bq,
                                                            (short)0, acc, false, false);
            }
            #pragma unroll
            for (int r = 0; r < 8; ++r) {
                const int tt = ti * 16 + (lane >> 4) * 8 + r;
                const size_t vidx = ((size_t)(b * T + t0 + tt) * NHEADS + h) * DVH + col;
                sU[tt * DVH + col] = sBeta[tt] * (v[vidx] - acc[r]);
            }
        }
        __syncthreads();

        // -- stage 4: forward substitution (I+M) u = rhs, per column --
        if (tid < DVH) {
            const int col = tid;
            for (int t = 1; t < CHUNKT; ++t) {
                float accv = 0.f;
                for (int s = 0; s < t; ++s)
                    accv = fmaf(sA[t * CHUNKT + s], sU[s * DVH + col], accv);
                sU[t * DVH + col] -= accv;
            }
        }
        __syncthreads();

        // -- stage 5: attn = tril_inc(Qt @ Kb^T) -> sA (reuse) --
        kda_qk_mm(sQt, sKb, sA, nullptr, lane, wave);
        __syncthreads();

        // -- stage 6: O = Qt @ S + attn @ U  -> global --
        #pragma unroll
        for (int it = 0; it < 4; ++it) {
            const int tix = wave * 4 + it;
            const int ti = tix >> 3, tj = tix & 7;
            const int mrow = ti * 16 + (lane & 15);
            const int col  = tj * 16 + (lane & 15);
            v8f acc = {};
            for (int kk = 0; kk < DKH; kk += 4) {           // Qt @ S
                const int k2 = kk + (lane >> 4) * 2;
                v2f a, bq;
                a.x  = sQt[mrow * DKH + k2];  a.y  = sQt[mrow * DKH + k2 + 1];
                bq.x = sS[k2 * DVH + col];    bq.y = sS[(k2 + 1) * DVH + col];
                acc = __builtin_amdgcn_wmma_f32_16x16x4_f32(false, a, false, bq,
                                                            (short)0, acc, false, false);
            }
            for (int kk = 0; kk < CHUNKT; kk += 4) {        // attn @ U
                const int k2 = kk + (lane >> 4) * 2;
                v2f a, bq;
                a.x  = sA[mrow * CHUNKT + k2];  a.y  = sA[mrow * CHUNKT + k2 + 1];
                bq.x = sU[k2 * DVH + col];      bq.y = sU[(k2 + 1) * DVH + col];
                acc = __builtin_amdgcn_wmma_f32_16x16x4_f32(false, a, false, bq,
                                                            (short)0, acc, false, false);
            }
            #pragma unroll
            for (int r = 0; r < 8; ++r) {
                const int tt = ti * 16 + (lane >> 4) * 8 + r;
                o[((size_t)(b * T + t0 + tt) * NHEADS + h) * DVH + col] = acc[r];
            }
        }
        __syncthreads();

        // -- stage 7: S = diag(eb_last) S + Kd^T @ U, Kd[t][d]=Kb[t][d]*eb_last[d] --
        #pragma unroll
        for (int it = 0; it < 8; ++it) {
            const int tix = wave * 8 + it;
            const int ti = tix >> 3, tj = tix & 7;
            const int mrow = ti * 16 + (lane & 15);     // d index
            const int col  = tj * 16 + (lane & 15);     // v index
            const float ebm = sEb[mrow];
            v8f acc = {};
            for (int kk = 0; kk < CHUNKT; kk += 4) {
                const int k2 = kk + (lane >> 4) * 2;    // t index
                v2f a, bq;
                a.x  = sKb[k2 * DKH + mrow] * ebm;
                a.y  = sKb[(k2 + 1) * DKH + mrow] * ebm;
                bq.x = sU[k2 * DVH + col];
                bq.y = sU[(k2 + 1) * DVH + col];
                acc = __builtin_amdgcn_wmma_f32_16x16x4_f32(false, a, false, bq,
                                                            (short)0, acc, false, false);
            }
            #pragma unroll
            for (int r = 0; r < 8; ++r) {
                const int dd = ti * 16 + (lane >> 4) * 8 + r;
                sS[dd * DVH + col] = sEb[dd] * sS[dd * DVH + col] + acc[r];
            }
        }
        __syncthreads();
    }
}

// ===========================================================================
// RMS-norm over DV + rms_w + sigmoid(gate). One wave per (b,t,h); b128 in,
// bf16 out (feeds the final WMMA GEMM directly).
// ===========================================================================
__global__ __launch_bounds__(256)
void rms_gate(const float* __restrict__ o, const float* __restrict__ gate_pre,
              const float* __restrict__ rmsw, __bf16* __restrict__ out,
              int Bsz, int T)
{
    const int gw   = (blockIdx.x * blockDim.x + threadIdx.x) >> 5;
    const int lane = threadIdx.x & 31;
    if (gw >= Bsz * T * NHEADS) return;
    const int h  = gw % NHEADS;
    const int bt = gw / NHEADS;
    const size_t base = (size_t)bt * NHEADS * DVH + (size_t)h * DVH + lane * 4;

    float4 ov = *(const float4*)&o[base];
    float ss = ov.x * ov.x + ov.y * ov.y + ov.z * ov.z + ov.w * ov.w;
    #pragma unroll
    for (int off = 16; off > 0; off >>= 1) ss += __shfl_xor(ss, off, 32);
    const float r = rsqrtf(ss * (1.f / DVH) + 1e-6f);

    float4 gv = *(const float4*)&gate_pre[base];
    float4 rw = *(const float4*)&rmsw[lane * 4];
    bf16x4 res = { (__bf16)(ov.x * r * rw.x * sigmoid_(gv.x)),
                   (__bf16)(ov.y * r * rw.y * sigmoid_(gv.y)),
                   (__bf16)(ov.z * r * rw.z * sigmoid_(gv.z)),
                   (__bf16)(ov.w * r * rw.w * sigmoid_(gv.w)) };
    *(bf16x4*)&out[base] = res;
}

// ===========================================================================
// Host-side orchestration
// ===========================================================================
static inline void launch_gemm(const __bf16* A, const __bf16* B, float* C,
                               int M, int N, int K, hipStream_t s)
{
    dim3 grid((N + BN - 1) / BN, (M + BM - 1) / BM);
    gemm_bf16_wmma<false><<<grid, dim3(256), 0, s>>>(A, B, C, M, N, K);
}
static inline void launch_gemm_ng(const __bf16* A, const __bf16* B, float* C,
                                  int M, int N, int K, hipStream_t s)
{
    dim3 grid((N + BN - 1) / BN, (M + BM - 1) / BM);
    gemm_bf16_wmma<true><<<grid, dim3(256), 0, s>>>(A, B, C, M, N, K);
}
static inline void launch_cvt(const float* in, __bf16* out, int n, hipStream_t s)
{
    to_bf16<<<(n / 4 + 255) / 256, 256, 0, s>>>(in, out, n);
}

extern "C" void kernel_launch(void* const* d_in, const int* in_sizes, int n_in,
                              void* d_out, int out_size, void* d_ws, size_t ws_size,
                              hipStream_t stream)
{
    const float* x    = (const float*)d_in[0];
    const float* Wq   = (const float*)d_in[1];
    const float* Wk   = (const float*)d_in[2];
    const float* Wv   = (const float*)d_in[3];
    const float* Wo   = (const float*)d_in[4];
    const float* Wad  = (const float*)d_in[5];
    const float* Wau  = (const float*)d_in[6];
    const float* Wb   = (const float*)d_in[7];
    const float* Wgd  = (const float*)d_in[8];
    const float* Wgu  = (const float*)d_in[9];
    const float* cqw  = (const float*)d_in[10];
    const float* ckw  = (const float*)d_in[11];
    const float* cvw  = (const float*)d_in[12];
    const float* rmsw = (const float*)d_in[13];
    float* out = (float*)d_out;

    const int T  = 4096;
    const int BT = in_sizes[0] / D_MODEL;      // B*T
    const int Bsz = BT / T;
    const int M  = BT;

    // ---- workspace layout: fp32 region then bf16 region ----
    const size_t BIG = (size_t)M * D_MODEL;
    const size_t LOW = (size_t)M * DKH;
    const size_t W22 = (size_t)D_MODEL * D_MODEL;
    const size_t WLR = (size_t)DKH * D_MODEL;  // 128x2048 (and 2048x128)

    float* ws   = (float*)d_ws;
    float* bufA = ws;                          // GEMM C scratch / g
    float* bufQ = bufA + BIG;                  // q (later gate_pre)
    float* bufK = bufQ + BIG;                  // k
    float* bufV = bufK + BIG;                  // v
    float* bufO = bufV + BIG;                  // scan output
    float* bufL = bufO + BIG;                  // low-rank intermediate (fp32)
    float* bufB = bufL + LOW;                  // beta

    __bf16* hp  = (__bf16*)(bufB + (size_t)M * NHEADS);
    __bf16* xh   = hp;  hp += BIG;
    __bf16* Wqh  = hp;  hp += W22;
    __bf16* Wkh  = hp;  hp += W22;
    __bf16* Wvh  = hp;  hp += W22;
    __bf16* Woh  = hp;  hp += W22;
    __bf16* Wauh = hp;  hp += WLR;
    __bf16* Wguh = hp;  hp += WLR;
    __bf16* Wadh = hp;  hp += WLR;
    __bf16* Wgdh = hp;  hp += WLR;
    __bf16* Wbh  = hp;  hp += (size_t)D_MODEL * NHEADS;
    __bf16* adh  = hp;  hp += LOW;             // low-rank intermediate (bf16)
    __bf16* ooh  = hp;  hp += BIG;             // gated output (bf16)

    const int nBTH_blocks = (Bsz * T * NHEADS * 32 + 255) / 256;
    const int nBig = (int)BIG;

    // ---- one-time bf16 conversions (bandwidth-trivial) ----
    launch_cvt(x,   xh,   nBig, stream);
    launch_cvt(Wq,  Wqh,  (int)W22, stream);
    launch_cvt(Wk,  Wkh,  (int)W22, stream);
    launch_cvt(Wv,  Wvh,  (int)W22, stream);
    launch_cvt(Wo,  Woh,  (int)W22, stream);
    launch_cvt(Wau, Wauh, (int)WLR, stream);
    launch_cvt(Wgu, Wguh, (int)WLR, stream);
    launch_cvt(Wad, Wadh, (int)WLR, stream);
    launch_cvt(Wgd, Wgdh, (int)WLR, stream);
    launch_cvt(Wb,  Wbh,  D_MODEL * NHEADS, stream);

    // ---- q / k / v projections + conv/SiLU(/L2) ----
    launch_gemm(xh, Wqh, bufA, M, D_MODEL, D_MODEL, stream);
    conv_silu_norm<<<nBTH_blocks, 256, 0, stream>>>(bufA, cqw, bufQ, Bsz, T, 1);
    launch_gemm(xh, Wkh, bufA, M, D_MODEL, D_MODEL, stream);
    conv_silu_norm<<<nBTH_blocks, 256, 0, stream>>>(bufA, ckw, bufK, Bsz, T, 1);
    launch_gemm(xh, Wvh, bufA, M, D_MODEL, D_MODEL, stream);
    conv_silu_norm<<<nBTH_blocks, 256, 0, stream>>>(bufA, cvw, bufV, Bsz, T, 0);

    // ---- g = -softplus(clip(x @ Wad @ Wau)) ----
    launch_gemm(xh, Wadh, bufL, M, DKH, D_MODEL, stream);
    launch_cvt(bufL, adh, (int)LOW, stream);
    launch_gemm(adh, Wauh, bufA, M, D_MODEL, DKH, stream);
    ew_neg_softplus<<<(nBig + 255) / 256, 256, 0, stream>>>(bufA, nBig);

    // ---- beta = sigmoid(clip(x @ Wb))  (N=16 -> guarded GEMM) ----
    launch_gemm_ng(xh, Wbh, bufB, M, NHEADS, D_MODEL, stream);
    ew_sigmoid_clip<<<(M * NHEADS + 255) / 256, 256, 0, stream>>>(bufB, M * NHEADS);

    // ---- chunked delta-rule scan (one WGP per (b,h); ~209 KB dynamic LDS) ----
    const size_t smem = (size_t)(DKH * DVH + 4 * CHUNKT * DKH +
                                 CHUNKT * CHUNKT + DKH + CHUNKT) * sizeof(float);
    kda_chunk_scan<<<dim3(Bsz * NHEADS), dim3(256), smem, stream>>>(
        bufQ, bufK, bufV, bufA, bufB, bufO, Bsz, T);

    // ---- gate = sigmoid(x @ Wgd @ Wgu)  (bufQ free now) ----
    launch_gemm(xh, Wgdh, bufL, M, DVH, D_MODEL, stream);
    launch_cvt(bufL, adh, (int)LOW, stream);
    launch_gemm(adh, Wguh, bufQ, M, D_MODEL, DKH, stream);

    // ---- RMS norm + gate (bf16 out) -> output projection ----
    rms_gate<<<nBTH_blocks, 256, 0, stream>>>(bufO, bufQ, rmsw, ooh, Bsz, T);
    launch_gemm(ooh, Woh, out, M, D_MODEL, D_MODEL, stream);
}
